// SlotGateAttention_49271864820392
// MI455X (gfx1250) — matrix operand order repaired and checked by reading, not compiled, not run on hardware
//
#include <hip/hip_runtime.h>
#include <hip/hip_bf16.h>

// SlotGateAttention for MI455X (gfx1250, wave32).
// B=8, T=256, D=A=512.  fp32 in/out, bf16 WMMA (f32 accum) for the GEMMs.

typedef __attribute__((ext_vector_type(16))) __bf16 v16bf;
typedef __attribute__((ext_vector_type(8)))  __bf16 v8bf;
typedef __attribute__((ext_vector_type(8)))  float  v8f;

#define BB 8
#define TT 256
#define DD 512
#define AA 512

__device__ __forceinline__ unsigned short f2bf(float f) {
    // round-to-nearest-even fp32 -> bf16 bits
    unsigned u = __builtin_bit_cast(unsigned, f);
    u += 0x7FFFu + ((u >> 16) & 1u);
    return (unsigned short)(u >> 16);
}
__device__ __forceinline__ unsigned pack2bf(float lo, float hi) {
    return (unsigned)f2bf(lo) | ((unsigned)f2bf(hi) << 16);
}
__device__ __forceinline__ v16bf cat8(v8bf lo, v8bf hi) {
    return __builtin_shufflevector(lo, hi, 0, 1, 2, 3, 4, 5, 6, 7,
                                   8, 9, 10, 11, 12, 13, 14, 15);
}

// ---------------------------------------------------------------------------
// 128x64-tile GEMM:  C[bm:bm+128, bn:bn+64] = A(MxK) * Bm(KxN) (+bias)
// 256 threads = 8 waves; wave w owns a 32x32 output tile
//   rows [32*(w>>1), +32), cols [32*(w&1), +32)
// = 4 WMMA accumulators fed by 2 A-frags x 2 B-frags (each reused twice).
// A staged row-major bf16 in LDS (rows padded to 80 B so every fragment slice
// is a b128 load); B staged TRANSPOSED so B fragments are contiguous too.
// ---------------------------------------------------------------------------
template <bool HAS_BIAS>
__device__ void gemm_bf16_tile(const float* __restrict__ Aq,
                               const float* __restrict__ Bm,
                               const float* __restrict__ bias,
                               float* __restrict__ C,
                               int lda, int ldb, int ldc, int K,
                               int bm, int bn)
{
    __shared__ __align__(16) unsigned short lA [128][40]; // A tile 128x32 bf16
    __shared__ __align__(16) unsigned short lBt[64][40];  // B tile transposed [n][k]

    const int tid  = threadIdx.x;
    const int lane = tid & 31;
    const int w    = tid >> 5;
    const int wr   = w >> 1;        // 0..3 : 32-row strip
    const int wc   = w & 1;         // 0..1 : 32-col strip
    const int half = lane >> 4;     // 0: lanes 0-15, 1: lanes 16-31
    const int l16  = lane & 15;

    v8f acc00 = {}, acc01 = {}, acc10 = {}, acc11 = {};

    for (int k0 = 0; k0 < K; k0 += 32) {
        // ---- stage A tile 128x32 (16 elems/thread: 4x float4 -> 8x u32) ----
        {
            const int r = tid >> 1;
            const int c = (tid & 1) * 16;
            const float4* src =
                (const float4*)(Aq + (long)(bm + r) * lda + k0 + c);
            const float4 x0 = src[0];
            const float4 x1 = src[1];
            const float4 x2 = src[2];
            const float4 x3 = src[3];
            unsigned* dst = (unsigned*)&lA[r][c];
            dst[0] = pack2bf(x0.x, x0.y);
            dst[1] = pack2bf(x0.z, x0.w);
            dst[2] = pack2bf(x1.x, x1.y);
            dst[3] = pack2bf(x1.z, x1.w);
            dst[4] = pack2bf(x2.x, x2.y);
            dst[5] = pack2bf(x2.z, x2.w);
            dst[6] = pack2bf(x3.x, x3.y);
            dst[7] = pack2bf(x3.z, x3.w);
        }
        // ---- stage B tile 32x64 transposed (8 elems/thread) ----
        {
            const int r = tid >> 3;        // k row 0..31
            const int c = (tid & 7) * 8;   // n base
            const float4* src =
                (const float4*)(Bm + (long)(k0 + r) * ldb + bn + c);
            const float4 x0 = src[0];
            const float4 x1 = src[1];
            lBt[c + 0][r] = f2bf(x0.x);
            lBt[c + 1][r] = f2bf(x0.y);
            lBt[c + 2][r] = f2bf(x0.z);
            lBt[c + 3][r] = f2bf(x0.w);
            lBt[c + 4][r] = f2bf(x1.x);
            lBt[c + 5][r] = f2bf(x1.y);
            lBt[c + 6][r] = f2bf(x1.z);
            lBt[c + 7][r] = f2bf(x1.w);
        }
        __syncthreads();

        // ---- A fragments (ISA 7.12.2 16-bit A 16x32): lanes 0-15 hold
        //      K 0-7 & 16-23; lanes 16-31 hold K 8-15 & 24-31. b128 loads. ----
        const int kb  = half * 8;
        const int ar0 = wr * 32 + l16;
        const v16bf af0 = cat8(*(const v8bf*)&lA[ar0][kb],
                               *(const v8bf*)&lA[ar0][16 + kb]);
        const v16bf af1 = cat8(*(const v8bf*)&lA[ar0 + 16][kb],
                               *(const v8bf*)&lA[ar0 + 16][16 + kb]);

        // ---- B fragments: lanes 0-15 hold K 0-7 & 16-23; lanes 16-31 hold
        //      K 8-15 & 24-31; N = lane%16.  Contiguous in lBt -> b128 loads. ----
        const int nc0 = wc * 32 + l16;
        const v16bf bf0 = cat8(*(const v8bf*)&lBt[nc0][half * 8],
                               *(const v8bf*)&lBt[nc0][16 + half * 8]);
        const v16bf bf1 = cat8(*(const v8bf*)&lBt[nc0 + 16][half * 8],
                               *(const v8bf*)&lBt[nc0 + 16][16 + half * 8]);

        acc00 = __builtin_amdgcn_wmma_f32_16x16x32_bf16(
            false, af0, false, bf0, (short)0, acc00, false, false);
        acc01 = __builtin_amdgcn_wmma_f32_16x16x32_bf16(
            false, af0, false, bf1, (short)0, acc01, false, false);
        acc10 = __builtin_amdgcn_wmma_f32_16x16x32_bf16(
            false, af1, false, bf0, (short)0, acc10, false, false);
        acc11 = __builtin_amdgcn_wmma_f32_16x16x32_bf16(
            false, af1, false, bf1, (short)0, acc11, false, false);
        __syncthreads();
    }

    // C layout: VGPR r -> M=r (lanes 0-15) / M=8+r (lanes 16-31); N = lane%16
    const int col0 = bn + wc * 32 + l16;
    float b0 = 0.0f, b1 = 0.0f;
    if constexpr (HAS_BIAS) {    // hoisted: independent of r
        b0 = bias[col0];
        b1 = bias[col0 + 16];
    }
    const int row0 = bm + wr * 32 + half * 8;
#pragma unroll
    for (int r = 0; r < 8; ++r) {
        float* c0 = C + (long)(row0 + r) * ldc;
        c0[col0]      = acc00[r] + b0;
        c0[col0 + 16] = acc01[r] + b1;
        float* c1 = C + (long)(row0 + 16 + r) * ldc;
        c1[col0]      = acc10[r] + b0;
        c1[col0 + 16] = acc11[r] + b1;
    }
}

// ---------------------------------------------------------------------------
// K0: ii[b,a] = final_state[b,:] . W_ii[:,a] + b_ii[a]
// ---------------------------------------------------------------------------
__global__ __launch_bounds__(256)
void ii_kernel(const float* __restrict__ fs, const float* __restrict__ W,
               const float* __restrict__ bias, float* __restrict__ ii)
{
    const int idx = blockIdx.x * 256 + threadIdx.x;     // 0..4095
    const int b = idx >> 9;
    const int a = idx & 511;
    float s = bias[a];
    const float* f = fs + (long)b * DD;
    for (int d = 0; d < DD; ++d) s += f[d] * W[(long)d * AA + a];
    ii[idx] = s;
}

// ---------------------------------------------------------------------------
// K1: ih/sh/si = state(2048x512) @ {W_ih,W_sh,W_si} + bias   (blockIdx.z picks)
// ---------------------------------------------------------------------------
__global__ __launch_bounds__(256)
void proj_gemm_kernel(const float* __restrict__ X,
                      const float* __restrict__ W0, const float* __restrict__ b0,
                      const float* __restrict__ W1, const float* __restrict__ b1,
                      const float* __restrict__ W2, const float* __restrict__ b2,
                      float* __restrict__ Y0, float* __restrict__ Y1,
                      float* __restrict__ Y2)
{
    const float* W; const float* bias; float* Y;
    switch (blockIdx.z) {
        case 0:  W = W0; bias = b0; Y = Y0; break;
        case 1:  W = W1; bias = b1; Y = Y1; break;
        default: W = W2; bias = b2; Y = Y2; break;
    }
    gemm_bf16_tile<true>(X, W, bias, Y, DD, AA, AA, DD,
                         blockIdx.x * 128, blockIdx.y * 64);
}

// ---------------------------------------------------------------------------
// K2: intent score -> softmax over T -> c_intent   (one block per batch)
// ---------------------------------------------------------------------------
__global__ __launch_bounds__(256)
void intent_kernel(const float* __restrict__ ih, const float* __restrict__ iiv,
                   const float* __restrict__ v,  const float* __restrict__ state,
                   float* __restrict__ out_ci)
{
    const int b = blockIdx.x;
    const int t = threadIdx.x;
    const float* ihp = ih  + ((long)b * TT + t) * AA;
    const float* iip = iiv + (long)b * AA;
    float s = 0.0f;
    for (int a = 0; a < AA; ++a) s += v[a] * tanhf(ihp[a] + iip[a]);

    __shared__ float red[256];
    __shared__ float wts[256];
    red[t] = s; __syncthreads();
    for (int off = 128; off > 0; off >>= 1) {
        if (t < off) red[t] = fmaxf(red[t], red[t + off]);
        __syncthreads();
    }
    const float m = red[0]; __syncthreads();
    const float e = __expf(s - m);
    red[t] = e; __syncthreads();
    for (int off = 128; off > 0; off >>= 1) {
        if (t < off) red[t] += red[t + off];
        __syncthreads();
    }
    const float inv = 1.0f / red[0];
    __syncthreads();
    wts[t] = e * inv; __syncthreads();

    for (int d = t; d < DD; d += 256) {
        float acc = 0.0f;
        for (int s2 = 0; s2 < TT; ++s2)
            acc += wts[s2] * state[((long)b * TT + s2) * DD + d];
        out_ci[(long)b * DD + d] = acc;
    }
}

// ---------------------------------------------------------------------------
// K3: slot_score[b,t,s] = sum_a v[a]*tanh(sh[b,s,a]+si[b,t,a]); softmax over s
// one block per (t,b); lane s; si row staged in LDS. tanh-bound (TRANS VALU).
// ---------------------------------------------------------------------------
__global__ __launch_bounds__(256)
void slot_score_kernel(const float* __restrict__ sh, const float* __restrict__ si,
                       const float* __restrict__ v,  float* __restrict__ slot_w)
{
    const int t = blockIdx.x;
    const int b = blockIdx.y;
    const int s = threadIdx.x;

    __shared__ float q[AA];
    const float* sip = si + ((long)b * TT + t) * AA;
    for (int a = threadIdx.x; a < AA; a += 256) q[a] = sip[a];
    __syncthreads();

    const float* key = sh + ((long)b * TT + s) * AA;
    __builtin_prefetch(key, 0, 0);
    float sc = 0.0f;
    for (int a = 0; a < AA; ++a) sc += v[a] * tanhf(key[a] + q[a]);

    __shared__ float red[256];
    red[s] = sc; __syncthreads();
    for (int off = 128; off > 0; off >>= 1) {
        if (s < off) red[s] = fmaxf(red[s], red[s + off]);
        __syncthreads();
    }
    const float m = red[0]; __syncthreads();
    const float e = __expf(sc - m);
    red[s] = e; __syncthreads();
    for (int off = 128; off > 0; off >>= 1) {
        if (s < off) red[s] += red[s + off];
        __syncthreads();
    }
    slot_w[((long)b * TT + t) * TT + s] = e / red[0];
}

// ---------------------------------------------------------------------------
// K4: c_slot[b] = slot_w[b](256x256) @ state[b](256x512)   (bf16 WMMA)
// ---------------------------------------------------------------------------
__global__ __launch_bounds__(256)
void apply_gemm_kernel(const float* __restrict__ slot_w,
                       const float* __restrict__ state,
                       float* __restrict__ out)
{
    const int b = blockIdx.z;
    gemm_bf16_tile<false>(slot_w + (long)b * TT * TT,
                          state  + (long)b * TT * DD,
                          nullptr,
                          out    + (long)b * TT * DD,
                          TT, DD, DD, TT,
                          blockIdx.x * 128, blockIdx.y * 64);
}

// ---------------------------------------------------------------------------
extern "C" void kernel_launch(void* const* d_in, const int* in_sizes, int n_in,
                              void* d_out, int out_size, void* d_ws, size_t ws_size,
                              hipStream_t stream)
{
    const float* state   = (const float*)d_in[0];
    const float* final_s = (const float*)d_in[1];
    const float* intentv = (const float*)d_in[2];
    const float* W_ih    = (const float*)d_in[3];
    const float* b_ih    = (const float*)d_in[4];
    const float* W_ii    = (const float*)d_in[5];
    const float* b_ii    = (const float*)d_in[6];
    const float* slotv   = (const float*)d_in[7];
    const float* W_sh    = (const float*)d_in[8];
    const float* b_sh    = (const float*)d_in[9];
    const float* W_si    = (const float*)d_in[10];
    const float* b_si    = (const float*)d_in[11];

    float* ws     = (float*)d_ws;
    float* ih     = ws;                               // B*T*A  = 1,048,576
    float* sh     = ih + (long)BB * TT * AA;          // B*T*A
    float* si     = sh + (long)BB * TT * AA;          // B*T*A
    float* ii     = si + (long)BB * TT * AA;          // B*A    = 4096
    float* slot_w = ii + (long)BB * AA;               // B*T*T  = 524,288

    float* c_slot   = (float*)d_out;                  // B*T*D
    float* c_intent = c_slot + (long)BB * TT * DD;    // B*D

    // K0: ii
    ii_kernel<<<(BB * AA) / 256, 256, 0, stream>>>(final_s, W_ii, b_ii, ii);

    // K1: ih, sh, si projections (bf16 WMMA)
    dim3 g1((BB * TT) / 128, AA / 64, 3);
    proj_gemm_kernel<<<g1, 256, 0, stream>>>(state, W_ih, b_ih, W_sh, b_sh,
                                             W_si, b_si, ih, sh, si);

    // K2: intent attention -> c_intent
    intent_kernel<<<BB, 256, 0, stream>>>(ih, ii, intentv, state, c_intent);

    // K3: pairwise slot scores + softmax
    slot_score_kernel<<<dim3(TT, BB), 256, 0, stream>>>(sh, si, slotv, slot_w);

    // K4: c_slot = slot_w @ state (bf16 WMMA, batched over B)
    apply_gemm_kernel<<<dim3(TT / 128, DD / 64, BB), 256, 0, stream>>>(
        slot_w, state, c_slot);
}
